// NeuralODEDynamics_9328668966991
// MI455X (gfx1250) — matrix operand order repaired and checked by reading, not compiled, not run on hardware
//
#include <hip/hip_runtime.h>
#include <math.h>

#define LATENT   128
#define HIDDEN   256
#define TDIM     32
#define K0       160      // LATENT + TDIM
#define K1       288      // HIDDEN + TDIM
#define SSTR     296      // staging row stride in f16 elements (K1 + pad)
#define NROWS    16       // batch rows per workgroup (one WMMA M tile)
#define NWAVES   8
#define NTHREADS 256
#define LN_EPS   1e-5f

typedef __attribute__((ext_vector_type(16))) _Float16 v16h;
typedef __attribute__((ext_vector_type(8)))  float    v8f;

__device__ __forceinline__ v8f wmma_f16(v16h a, v16h b, v8f c) {
  // D = A(16x32 f16) * B(32x16 f16) + C(16x16 f32)
  return __builtin_amdgcn_wmma_f32_16x16x32_f16(false, a, false, b, (short)0, c, false, false);
}

// A fragment (16x32 f16) from row-major LDS staging [m][k], stride SSTR.
// ISA layout: lanes 0-15: M=lane, VGPR0-3 K=0..7, VGPR4-7 K=16..23;
//             lanes 16-31: same M set, K ranges +8.
__device__ __forceinline__ v16h ldA(const _Float16* stg, int m, int ks, int hi) {
  const _Float16* p = stg + m * SSTR + ks + 8 * hi;
  union { int4 i[2]; v16h h; } u;
  u.i[0] = *(const int4*)p;          // K = ks + 8*hi + 0..7
  u.i[1] = *(const int4*)(p + 16);   // K = ks + 16 + 8*hi + 0..7
  return u.h;
}

// B fragment (32x16 f16) from transposed weights in LDS: wT[n][k], stride Kstr.
// ISA layout: lanes 0-15: N=lane, K=0..7,16..23; lanes 16-31: N=lane-16, K=8..15,24..31.
__device__ __forceinline__ v16h ldB(const _Float16* wT, int Kstr, int n, int ks, int hi) {
  const _Float16* p = wT + n * Kstr + ks + 8 * hi;
  union { int4 i[2]; v16h h; } u;
  u.i[0] = *(const int4*)p;
  u.i[1] = *(const int4*)(p + 16);
  return u.h;
}

// softplus via hardware TRANS ops (v_exp_f32 / v_log_f32), numerically stable.
__device__ __forceinline__ float softplusf(float x) {
  float ax = fabsf(x);
  float e  = __expf(-ax);              // in (0,1]
  float l  = __logf(1.f + e);          // log1p(exp(-|x|))
  return (x > 0.f) ? (x + l) : l;
}

// GEMM (K=KTOT) into 2 N-tiles per wave, + bias + LayerNorm + softplus,
// result written as f16 into staging cols [32*wv, 32*wv+32).
template <int KTOT>
__device__ __forceinline__ void gemm_ln(const _Float16* __restrict__ wT,
                                        _Float16* __restrict__ stg,
                                        float* __restrict__ stats,
                                        int wv, int lm, int hi,
                                        float bc0, float bc1,
                                        float g0, float g1, float bb0, float bb1) {
  v8f a0 = {0.f,0.f,0.f,0.f,0.f,0.f,0.f,0.f};
  v8f a1 = {0.f,0.f,0.f,0.f,0.f,0.f,0.f,0.f};
#pragma unroll
  for (int ks = 0; ks < KTOT; ks += 32) {
    v16h af  = ldA(stg, lm, ks, hi);
    v16h bf0 = ldB(wT, KTOT, 32 * wv + lm,      ks, hi);
    v16h bf1 = ldB(wT, KTOT, 32 * wv + 16 + lm, ks, hi);
    a0 = wmma_f16(af, bf0, a0);
    a1 = wmma_f16(af, bf1, a1);
  }
  // C layout: vgpr j, lanes 0-15 -> row j, col=lane; lanes 16-31 -> row j+8.
  float v0[8], v1[8], p[8], q[8];
#pragma unroll
  for (int j = 0; j < 8; ++j) {
    v0[j] = a0[j] + bc0;
    v1[j] = a1[j] + bc1;
    p[j] = v0[j] + v1[j];
    q[j] = v0[j] * v0[j] + v1[j] * v1[j];
  }
  // Butterfly reduce across the 16 lanes holding one row-half (masks stay in-half).
#pragma unroll
  for (int m = 1; m < 16; m <<= 1) {
#pragma unroll
    for (int j = 0; j < 8; ++j) {
      p[j] += __shfl_xor(p[j], m, 32);
      q[j] += __shfl_xor(q[j], m, 32);
    }
  }
  if (lm == 0) {  // lanes 0 and 16 publish this wave's 32-col partials
#pragma unroll
    for (int j = 0; j < 8; ++j) {
      int row = hi * 8 + j;
      stats[(wv * 16 + row) * 2 + 0] = p[j];
      stats[(wv * 16 + row) * 2 + 1] = q[j];
    }
  }
  __syncthreads();  // stats ready; also: all waves done reading staging
#pragma unroll
  for (int j = 0; j < 8; ++j) {
    int row = hi * 8 + j;
    float s = 0.f, sq = 0.f;
#pragma unroll
    for (int w = 0; w < NWAVES; ++w) {
      s  += stats[(w * 16 + row) * 2 + 0];
      sq += stats[(w * 16 + row) * 2 + 1];
    }
    float mean = s * (1.0f / HIDDEN);
    float var  = sq * (1.0f / HIDDEN) - mean * mean;
    float rs   = rsqrtf(var + LN_EPS);
    float h0 = softplusf((v0[j] - mean) * rs * g0 + bb0);
    float h1 = softplusf((v1[j] - mean) * rs * g1 + bb1);
    stg[row * SSTR + 32 * wv + lm]      = (_Float16)h0;
    stg[row * SSTR + 32 * wv + 16 + lm] = (_Float16)h1;
  }
  __syncthreads();  // h staging ready for next GEMM
}

// Output layer: one 16-col N-tile per wave, k = C + b2 kept in registers.
template <int KTOT>
__device__ __forceinline__ void gemm_out(const _Float16* __restrict__ wT,
                                         const _Float16* __restrict__ stg,
                                         int wv, int lm, int hi, float b2c,
                                         float* __restrict__ kout) {
  v8f a0 = {0.f,0.f,0.f,0.f,0.f,0.f,0.f,0.f};
#pragma unroll
  for (int ks = 0; ks < KTOT; ks += 32) {
    v16h af = ldA(stg, lm, ks, hi);
    v16h bf = ldB(wT, KTOT, 16 * wv + lm, ks, hi);
    a0 = wmma_f16(af, bf, a0);
  }
  __syncthreads();  // all waves done reading staging -> caller may rebuild y
#pragma unroll
  for (int j = 0; j < 8; ++j) kout[j] = a0[j] + b2c;
}

// y staging write: wave owns cols [16*wv, 16*wv+16) of the 128 latent cols.
__device__ __forceinline__ void writeY(_Float16* __restrict__ stg,
                                       const float* z, const float* k, float c,
                                       int col, int hi) {
#pragma unroll
  for (int j = 0; j < 8; ++j)
    stg[(hi * 8 + j) * SSTR + col] = (_Float16)(z[j] + c * k[j]);
}

// One evaluation of the learned vector field f(t, y) -> kout (registers).
// Caller must have written y into staging cols 0..127 just before.
__device__ __forceinline__ void ode_f(float t, float* __restrict__ kout,
    const _Float16* __restrict__ w0T, const _Float16* __restrict__ w1T,
    const _Float16* __restrict__ w2T, _Float16* __restrict__ stg,
    float* __restrict__ stats, const float* __restrict__ projW,
    const float* __restrict__ projB, const float* __restrict__ freqL,
    int wv, int lane, int lm, int hi,
    float b0c0, float b0c1, float g00, float g01, float be00, float be01,
    float b1c0, float b1c1, float g10, float g11, float be10, float be11,
    float b2c) {
  if (wv == 0) {  // time embedding: 32 outputs, one per lane of wave 0
    float emb[32];
#pragma unroll
    for (int e = 0; e < 16; ++e) {
      float a = t * freqL[e];
      emb[e]      = __sinf(a);
      emb[16 + e] = __cosf(a);
    }
    float acc = projB[lane];
#pragma unroll
    for (int e = 0; e < 32; ++e) acc += emb[e] * projW[e * 32 + lane];
    _Float16 hv = (_Float16)acc;
    for (int r = 0; r < NROWS; ++r) {
      stg[r * SSTR + LATENT + lane] = hv;  // cols 128..159 (layer-0 concat)
      stg[r * SSTR + HIDDEN + lane] = hv;  // cols 256..287 (layer-1/2 concat)
    }
  }
  __syncthreads();  // y + temb staged
  gemm_ln<K0>(w0T, stg, stats, wv, lm, hi, b0c0, b0c1, g00, g01, be00, be01);
  gemm_ln<K1>(w1T, stg, stats, wv, lm, hi, b1c0, b1c1, g10, g11, be10, be11);
  gemm_out<K1>(w2T, stg, wv, lm, hi, b2c, kout);
}

__global__ __launch_bounds__(NTHREADS) void neural_ode_rk4_kernel(
    const float* __restrict__ z0, const float* __restrict__ t_eval,
    const float* __restrict__ freq, const float* __restrict__ proj_w,
    const float* __restrict__ proj_b,
    const float* __restrict__ w0, const float* __restrict__ b0,
    const float* __restrict__ ln0_g, const float* __restrict__ ln0_b,
    const float* __restrict__ w1, const float* __restrict__ b1,
    const float* __restrict__ ln1_g, const float* __restrict__ ln1_b,
    const float* __restrict__ w2, const float* __restrict__ b2,
    float* __restrict__ out, int B, int T) {
  extern __shared__ char smemraw[];
  _Float16* w0T = (_Float16*)smemraw;        // [256][160] f16, transposed [n][k]
  _Float16* w1T = w0T + HIDDEN * K0;         // [256][288]
  _Float16* w2T = w1T + HIDDEN * K1;         // [128][288]
  _Float16* stg = w2T + LATENT * K1;         // [16][SSTR] activation staging
  float* stats  = (float*)(stg + NROWS * SSTR);  // [8][16][2]
  float* projW  = stats + NWAVES * 16 * 2;       // [32][32]
  float* projB  = projW + 32 * 32;               // [32]
  float* freqL  = projB + 32;                    // [16]

  const int tid  = threadIdx.x;
  const int wv   = tid >> 5;
  const int lane = tid & 31;
  const int lm   = lane & 15;
  const int hi   = lane >> 4;
  const int base = blockIdx.x * NROWS;

  // One-time: f16-transpose weights into LDS (shared by all 8 waves, all steps).
  for (int i = tid; i < K0 * HIDDEN; i += NTHREADS) {
    int k = i / HIDDEN, n = i % HIDDEN;
    w0T[n * K0 + k] = (_Float16)w0[i];
  }
  for (int i = tid; i < K1 * HIDDEN; i += NTHREADS) {
    int k = i / HIDDEN, n = i % HIDDEN;
    w1T[n * K1 + k] = (_Float16)w1[i];
  }
  for (int i = tid; i < K1 * LATENT; i += NTHREADS) {
    int k = i / LATENT, n = i % LATENT;
    w2T[n * K1 + k] = (_Float16)w2[i];
  }
  for (int i = tid; i < 32 * 32; i += NTHREADS) projW[i] = proj_w[i];
  if (tid < 32) projB[tid] = proj_b[tid];
  if (tid < 16) freqL[tid] = freq[tid];

  // Per-lane column constants (C-layout columns owned by this lane).
  const int c0 = 32 * wv + lm, c1 = c0 + 16, c2 = 16 * wv + lm;
  const float b0c0 = b0[c0], b0c1 = b0[c1];
  const float g00 = ln0_g[c0], g01 = ln0_g[c1], be00 = ln0_b[c0], be01 = ln0_b[c1];
  const float b1c0 = b1[c0], b1c1 = b1[c1];
  const float g10 = ln1_g[c0], g11 = ln1_g[c1], be10 = ln1_b[c0], be11 = ln1_b[c1];
  const float b2c = b2[c2];

  // RK4 state in registers, WMMA C-layout: rows hi*8+j, col c2.
  float z[8];
#pragma unroll
  for (int j = 0; j < 8; ++j) {
    int row = base + hi * 8 + j;
    z[j] = z0[(size_t)row * LATENT + c2];
    out[(size_t)row * LATENT + c2] = z[j];  // trajectory slice t=0
  }
  __syncthreads();

  for (int iv = 0; iv < T - 1; ++iv) {
    float ta   = t_eval[iv];
    float dtiv = t_eval[iv + 1] - ta;
    int n = (int)fabsf(dtiv);
    if (n < 1) n = 1;
    float sd = dtiv / (float)n;
    for (int s = 0; s < n; ++s) {
      float t = ta + (float)s * sd;
      float k1[8], k2[8], k3[8], k4[8];
      writeY(stg, z, z, 0.f, c2, hi);
      ode_f(t, k1, w0T, w1T, w2T, stg, stats, projW, projB, freqL,
            wv, lane, lm, hi, b0c0, b0c1, g00, g01, be00, be01,
            b1c0, b1c1, g10, g11, be10, be11, b2c);
      writeY(stg, z, k1, 0.5f * sd, c2, hi);
      ode_f(t + 0.5f * sd, k2, w0T, w1T, w2T, stg, stats, projW, projB, freqL,
            wv, lane, lm, hi, b0c0, b0c1, g00, g01, be00, be01,
            b1c0, b1c1, g10, g11, be10, be11, b2c);
      writeY(stg, z, k2, 0.5f * sd, c2, hi);
      ode_f(t + 0.5f * sd, k3, w0T, w1T, w2T, stg, stats, projW, projB, freqL,
            wv, lane, lm, hi, b0c0, b0c1, g00, g01, be00, be01,
            b1c0, b1c1, g10, g11, be10, be11, b2c);
      writeY(stg, z, k3, sd, c2, hi);
      ode_f(t + sd, k4, w0T, w1T, w2T, stg, stats, projW, projB, freqL,
            wv, lane, lm, hi, b0c0, b0c1, g00, g01, be00, be01,
            b1c0, b1c1, g10, g11, be10, be11, b2c);
      float w6 = sd * (1.0f / 6.0f);
#pragma unroll
      for (int j = 0; j < 8; ++j)
        z[j] += w6 * (k1[j] + 2.f * k2[j] + 2.f * k3[j] + k4[j]);
    }
#pragma unroll
    for (int j = 0; j < 8; ++j) {
      int row = base + hi * 8 + j;
      out[(size_t)(iv + 1) * B * LATENT + (size_t)row * LATENT + c2] = z[j];
    }
  }
}

static constexpr size_t SMEM_BYTES =
    (size_t)(HIDDEN * K0 + HIDDEN * K1 + LATENT * K1 + NROWS * SSTR) * sizeof(_Float16) +
    (size_t)(NWAVES * 16 * 2 + 32 * 32 + 32 + 16) * sizeof(float);  // = 317,888 B

extern "C" void kernel_launch(void* const* d_in, const int* in_sizes, int n_in,
                              void* d_out, int out_size, void* d_ws, size_t ws_size,
                              hipStream_t stream) {
  const float* z0     = (const float*)d_in[0];
  const float* t_eval = (const float*)d_in[1];
  const float* freq   = (const float*)d_in[2];
  const float* proj_w = (const float*)d_in[3];
  const float* proj_b = (const float*)d_in[4];
  const float* w0     = (const float*)d_in[5];
  const float* b0     = (const float*)d_in[6];
  const float* ln0_g  = (const float*)d_in[7];
  const float* ln0_b  = (const float*)d_in[8];
  const float* w1     = (const float*)d_in[9];
  const float* b1     = (const float*)d_in[10];
  const float* ln1_g  = (const float*)d_in[11];
  const float* ln1_b  = (const float*)d_in[12];
  const float* w2     = (const float*)d_in[13];
  const float* b2     = (const float*)d_in[14];
  float* out = (float*)d_out;

  int B = in_sizes[0] / LATENT;  // 4096
  int T = in_sizes[1];           // 8

  (void)hipFuncSetAttribute((const void*)neural_ode_rk4_kernel,
                            hipFuncAttributeMaxDynamicSharedMemorySize,
                            (int)SMEM_BYTES);

  dim3 grid(B / NROWS);     // 256 workgroups, 16 rows each
  dim3 block(NTHREADS);     // 8 waves
  hipLaunchKernelGGL(neural_ode_rk4_kernel, grid, block, SMEM_BYTES, stream,
                     z0, t_eval, freq, proj_w, proj_b,
                     w0, b0, ln0_g, ln0_b,
                     w1, b1, ln1_g, ln1_b,
                     w2, b2, out, B, T);
}
